// GCN32Diff_56444460204491
// MI455X (gfx1250) — matrix operand order repaired and checked by reading, not compiled, not run on hardware
//
#include <hip/hip_runtime.h>
#include <hip/hip_bf16.h>

typedef __attribute__((ext_vector_type(16))) _Float16 v16h;
typedef __attribute__((ext_vector_type(8)))  _Float16 half8;
typedef __attribute__((ext_vector_type(8)))  float    v8f;

#define B_   16
#define N_   2048
#define F_   128
#define HID_ 32
#define FC_  512
#define C_   10

#define WAVES 4                       // 4 waves * 16 rows = 64 rows per block
#define ROWS_PER_BLK (16*WAVES)
#define NBLK (N_/ROWS_PER_BLK)        // 32 row-blocks per batch

// ---------------------------------------------------------------------------
// Kernel 0: Xt[b][f][n] = (f16) X[b][n][f]   (B-operand needs K (=n) contiguous)
//           W1t[h][f]   = (f16) W1[f][h]     (B-operand of second GEMM)
// ---------------------------------------------------------------------------
__global__ __launch_bounds__(256)
void prep_kernel(const float* __restrict__ X, const float* __restrict__ W1,
                 _Float16* __restrict__ Xt, _Float16* __restrict__ W1t) {
    int idx = blockIdx.x * 256 + threadIdx.x;       // 0 .. B*F*N-1
    int n = idx & (N_ - 1);
    int f = (idx >> 11) & (F_ - 1);
    int b = idx >> 18;
    Xt[idx] = (_Float16)X[((size_t)b * N_ + n) * F_ + f];
    if (idx < HID_ * F_) {
        int h = idx / F_, ff = idx % F_;
        W1t[idx] = (_Float16)W1[ff * HID_ + h];
    }
}

// ---------------------------------------------------------------------------
// Kernel 1: fused  H = relu(A@X@W1 + b1)  ->  per-block partial column sums.
// One wave owns a 16-row M-tile; block = 4 waves = 64 rows of one batch.
// ---------------------------------------------------------------------------
__global__ __launch_bounds__(WAVES*32)
void gcn_main(const float* __restrict__ A,      // [B,N,N] fp32
              const _Float16* __restrict__ Xt,  // [B,F,N] f16
              const _Float16* __restrict__ W1t, // [HID,F] f16
              const float* __restrict__ b1,     // [HID]
              float* __restrict__ partial)      // [B,NBLK,HID]
{
    __shared__ __align__(16) _Float16 lds[WAVES][16][F_]; // AX tiles as f16
    __shared__ float red[WAVES][4][16];

    const int tid  = threadIdx.x;
    const int lane = tid & 31;
    const int wave = tid >> 5;
    const int b    = blockIdx.y;
    const int li   = lane & 15;
    const int hi   = lane >> 4;          // lane half
    const int koffA = hi ? 8 : 0;        // A frag: K {0..7,16..23} / {8..15,24..31}
    const int koffB = hi ? 8 : 0;        // B frag: K {0..7,16..23} / {8..15,24..31}
    const int m0   = blockIdx.x * ROWS_PER_BLK + wave * 16;

    const float*    Arow = A  + ((size_t)b * N_ + (m0 + li)) * (size_t)N_;
    const _Float16* Xb   = Xt + (size_t)b * F_ * N_;

    // ---- GEMM 1: AX tile (16 x 128), K = 2048 in steps of 32 ----
    v8f acc[8] = {};
    for (int kk = 0; kk < N_; kk += 32) {
        const float* a0 = Arow + kk + koffA;
        // Prefetch this lane's A stream 8 k-steps (1KB) ahead; speculative,
        // dropped silently if it runs past the mapping.
        __builtin_prefetch(a0 + 256, 0, 0);

        // Issue ALL fragment loads first so a single load clause stays in
        // flight while the wmma chain drains it with partial waits.
        float4 x0 = *(const float4*)(a0);
        float4 x1 = *(const float4*)(a0 + 4);
        float4 y0 = *(const float4*)(a0 + 16);
        float4 y1 = *(const float4*)(a0 + 20);

        half8 blo[8], bhi[8];
        #pragma unroll
        for (int t = 0; t < 8; ++t) {
            const _Float16* bp = Xb + (size_t)(t * 16 + li) * N_ + kk + koffB;
            blo[t] = *(const half8*)(bp);
            bhi[t] = *(const half8*)(bp + 16);
        }

        v16h af;
        af[0]=(_Float16)x0.x; af[1]=(_Float16)x0.y; af[2]=(_Float16)x0.z; af[3]=(_Float16)x0.w;
        af[4]=(_Float16)x1.x; af[5]=(_Float16)x1.y; af[6]=(_Float16)x1.z; af[7]=(_Float16)x1.w;
        af[8]=(_Float16)y0.x; af[9]=(_Float16)y0.y; af[10]=(_Float16)y0.z; af[11]=(_Float16)y0.w;
        af[12]=(_Float16)y1.x; af[13]=(_Float16)y1.y; af[14]=(_Float16)y1.z; af[15]=(_Float16)y1.w;

        #pragma unroll
        for (int t = 0; t < 8; ++t) {
            v16h bf = __builtin_shufflevector(blo[t], bhi[t],
                        0,1,2,3,4,5,6,7,8,9,10,11,12,13,14,15);
            acc[t] = __builtin_amdgcn_wmma_f32_16x16x32_f16(
                        false, af, false, bf, (short)0, acc[t], false, false);
        }
    }

    // ---- re-fragment AX through LDS (C-layout -> A-layout), as f16 ----
    #pragma unroll
    for (int t = 0; t < 8; ++t)
        #pragma unroll
        for (int j = 0; j < 8; ++j)
            lds[wave][hi ? j + 8 : j][t * 16 + li] = (_Float16)acc[t][j];
    __syncthreads();

    // ---- GEMM 2: H tile = AX(16x128) @ W1(128x32), K steps of 32 ----
    v8f acc2[2] = {};
    #pragma unroll
    for (int kk = 0; kk < F_; kk += 32) {
        const _Float16* ap = &lds[wave][li][kk + koffA];
        half8 alo = *(const half8*)(ap);
        half8 ahi = *(const half8*)(ap + 16);
        v16h af = __builtin_shufflevector(alo, ahi,
                    0,1,2,3,4,5,6,7,8,9,10,11,12,13,14,15);
        #pragma unroll
        for (int t = 0; t < 2; ++t) {
            const _Float16* bp = W1t + (size_t)(t * 16 + li) * F_ + kk + koffB;
            half8 blo = *(const half8*)(bp);
            half8 bhi = *(const half8*)(bp + 16);
            v16h bf = __builtin_shufflevector(blo, bhi,
                        0,1,2,3,4,5,6,7,8,9,10,11,12,13,14,15);
            acc2[t] = __builtin_amdgcn_wmma_f32_16x16x32_f16(
                        false, af, false, bf, (short)0, acc2[t], false, false);
        }
    }

    // ---- bias + relu + column partial sums (deterministic, fixed order) ----
    #pragma unroll
    for (int t = 0; t < 2; ++t) {
        int n = t * 16 + li;
        float bias = b1[n];
        float s = 0.f;
        #pragma unroll
        for (int j = 0; j < 8; ++j) s += fmaxf(acc2[t][j] + bias, 0.f);
        red[wave][t * 2 + hi][li] = s;
    }
    __syncthreads();
    if (tid < HID_) {
        int n = tid;
        float s = 0.f;
        #pragma unroll
        for (int w = 0; w < WAVES; ++w)
            s += red[w][(n >> 4) * 2 + 0][n & 15] + red[w][(n >> 4) * 2 + 1][n & 15];
        partial[((size_t)b * NBLK + blockIdx.x) * HID_ + n] = s;
    }
}

// ---------------------------------------------------------------------------
// Kernel 2: Xp = sum(partials); out = softmax(relu(Xp@Wd+bd)@Wc+bc). One block/batch.
// (S = softmax over a size-1 axis == 1, so Xp is just the column sum of H.)
// ---------------------------------------------------------------------------
__global__ __launch_bounds__(FC_)
void gcn_head(const float* __restrict__ partial,
              const float* __restrict__ Wd, const float* __restrict__ bd,
              const float* __restrict__ Wc, const float* __restrict__ bc,
              float* __restrict__ out)
{
    __shared__ float xs[HID_];
    __shared__ float hb[FC_];
    __shared__ float lg[C_];
    const int b = blockIdx.x, j = threadIdx.x;

    if (j < HID_) {
        float s = 0.f;
        for (int blk = 0; blk < NBLK; ++blk)
            s += partial[((size_t)b * NBLK + blk) * HID_ + j];
        xs[j] = s;
    }
    __syncthreads();

    float a = bd[j];
    #pragma unroll
    for (int k = 0; k < HID_; ++k) a += xs[k] * Wd[k * FC_ + j];
    hb[j] = fmaxf(a, 0.f);
    __syncthreads();

    if (j < C_) {
        float l = bc[j];
        for (int k = 0; k < FC_; ++k) l += hb[k] * Wc[k * C_ + j];
        lg[j] = l;
    }
    __syncthreads();

    if (j == 0) {
        float mx = lg[0];
        for (int c = 1; c < C_; ++c) mx = fmaxf(mx, lg[c]);
        float e[C_]; float sum = 0.f;
        for (int c = 0; c < C_; ++c) { e[c] = expf(lg[c] - mx); sum += e[c]; }
        float inv = 1.f / sum;
        for (int c = 0; c < C_; ++c) out[b * C_ + c] = e[c] * inv;
    }
}

// ---------------------------------------------------------------------------
extern "C" void kernel_launch(void* const* d_in, const int* in_sizes, int n_in,
                              void* d_out, int out_size, void* d_ws, size_t ws_size,
                              hipStream_t stream) {
    const float* filtre = (const float*)d_in[0];
    const float* X      = (const float*)d_in[1];
    // d_in[2] node_indicator: unused by reference
    const float* W1     = (const float*)d_in[3];
    const float* b1     = (const float*)d_in[4];
    // d_in[5] Ws: dead code (softmax over size-1 axis == 1)
    const float* Wd     = (const float*)d_in[6];
    const float* bd     = (const float*)d_in[7];
    const float* Wc     = (const float*)d_in[8];
    const float* bc     = (const float*)d_in[9];
    float* out = (float*)d_out;

    char* ws = (char*)d_ws;
    _Float16* Xt  = (_Float16*)ws;                                   // 8 MB
    _Float16* W1t = (_Float16*)(ws + (size_t)8 * 1024 * 1024);       // 8 KB
    float* partial = (float*)(ws + (size_t)8 * 1024 * 1024 + 8192);  // 64 KB

    prep_kernel<<<(B_ * F_ * N_) / 256, 256, 0, stream>>>(X, W1, Xt, W1t);
    gcn_main<<<dim3(NBLK, B_), WAVES * 32, 0, stream>>>(filtre, Xt, W1t, b1, partial);
    gcn_head<<<B_, FC_, 0, stream>>>(partial, Wd, bd, Wc, bc, out);
}
